// BitNetLinear4Bit_39668317946069
// MI455X (gfx1250) — matrix operand order repaired and checked by reading, not compiled, not run on hardware
//
#include <hip/hip_runtime.h>
#include <cstdint>

// ---------------------------------------------------------------------------
// BitNet ternary linear:  out[m,n] = alpha * sum_k x[m,k] * T[n,k] + bias[n]
//   T = clamp(round(w/alpha), -1, 1)  (exact in bf16; alpha folded to epilogue)
// M = 4096, N = 11008, K = 4096.
// GEMM: v_wmma_f32_16x16x32_bf16, block tile 128x256, wave tile 64x64,
//       LDS fed by the Tensor Data Mover (tensor_load_to_lds + TENSORcnt).
// ---------------------------------------------------------------------------

typedef __attribute__((ext_vector_type(16))) __bf16 v16bf;
typedef __attribute__((ext_vector_type(8)))  __bf16 v8bf;
typedef __attribute__((ext_vector_type(8)))  float  v8f;
typedef __attribute__((ext_vector_type(4)))  unsigned int u32x4;
typedef __attribute__((ext_vector_type(8)))  int i32x8;
typedef __attribute__((ext_vector_type(4)))  int i32x4;

constexpr int MD = 4096;      // B*S
constexpr int ND = 11008;     // DOUT
constexpr int KD = 4096;      // DIN

constexpr int BM  = 128;      // block tile M
constexpr int BN  = 256;      // block tile N
constexpr int BKT = 64;       // block tile K (2 wmma K-steps)
constexpr int LDT = BKT + 8;  // padded LDS row stride (bf16) = 72 (144 B)
constexpr int NT  = KD / BKT; // 64 K-tiles

#if defined(__has_builtin)
# if __has_builtin(__builtin_amdgcn_tensor_load_to_lds)
#  define USE_TDM 1
# endif
#endif

// -------------------------- prep kernels -----------------------------------

__global__ void cvt_x_bf16(const float* __restrict__ x,
                           __bf16* __restrict__ xq, int n8) {
  int i = blockIdx.x * blockDim.x + threadIdx.x;
  if (i >= n8) return;
  const float4* p = (const float4*)x + (size_t)i * 2;
  float4 a = p[0], b = p[1];
  v8bf o;
  o[0] = (__bf16)a.x; o[1] = (__bf16)a.y; o[2] = (__bf16)a.z; o[3] = (__bf16)a.w;
  o[4] = (__bf16)b.x; o[5] = (__bf16)b.y; o[6] = (__bf16)b.z; o[7] = (__bf16)b.w;
  *((v8bf*)xq + i) = o;
}

__global__ void quant_w_ternary(const float* __restrict__ w,
                                const float* __restrict__ alphap,
                                __bf16* __restrict__ wq, int n8) {
  int i = blockIdx.x * blockDim.x + threadIdx.x;
  if (i >= n8) return;
  const float a = alphap[0];
  const float4* p = (const float4*)w + (size_t)i * 2;
  float4 u = p[0], v = p[1];
  float t[8] = {u.x, u.y, u.z, u.w, v.x, v.y, v.z, v.w};
  v8bf o;
#pragma unroll
  for (int j = 0; j < 8; ++j) {
    float q = rintf(t[j] / a);            // round-half-even, matches jnp.round
    q = fminf(1.0f, fmaxf(-1.0f, q));
    o[j] = (__bf16)q;                     // {-1,0,1}: exact in bf16
  }
  *((v8bf*)wq + i) = o;
}

// -------------------------- fragment loads ---------------------------------

// A frag (16x32): lane l = row M=l%16; elems 0..7 = K[(l<16?0:8)..], 8..15 = +16
// B frag (32x16): lane l = col N=l%16; elems 0..7 = K[(l<16?0:8)..], 8..15 = +16
static __device__ __forceinline__ v16bf frag16(const __bf16* p, int hi_off) {
  v8bf lo = *(const v8bf*)p;
  v8bf hi = *(const v8bf*)(p + hi_off);
  return __builtin_shufflevector(lo, hi, 0, 1, 2, 3, 4, 5, 6, 7,
                                          8, 9, 10, 11, 12, 13, 14, 15);
}

#ifdef USE_TDM
// Issue a TDM 2D tile load: rows x 64 bf16 elements, row stride = KD elems,
// LDS padded by 16B every 128B (-> row stride 144B = LDT bf16).
// D# bitfields per CDNA5 ISA 8.3/8.4.
static __device__ __forceinline__ void tdm_load_2d(unsigned long long gaddr,
                                                   unsigned lds_addr,
                                                   unsigned rows) {
  u32x4 g0;
  g0[0] = 1u;                                        // count=1, user descriptor
  g0[1] = lds_addr;                                  // [63:32] lds_addr
  g0[2] = (unsigned)(gaddr & 0xFFFFFFFFu);           // [95:64] global_addr lo
  g0[3] = ((unsigned)(gaddr >> 32) & 0x01FFFFFFu)    // [120:96] global_addr hi
        | (2u << 30);                                // [127:126] type=2 (image)
  i32x8 g1;
  g1[0] = (int)((0u)                                 // [15:0] workgroup_mask=0
              | (1u << 16)                           // [17:16] data_size=2B
              | (1u << 20)                           // [20] pad_enable
              | (4u << 22)                           // [24:22] pad every 32 DW
              | (3u << 25));                         // [31:25] pad 4 DWORDs
  g1[1] = (int)(((unsigned)KD & 0xFFFFu) << 16);     // tensor_dim0 lo16 (4096)
  g1[2] = (int)((((unsigned)KD >> 16) & 0xFFFFu)     // tensor_dim0 hi16
              | ((rows & 0xFFFFu) << 16));           // tensor_dim1 lo16
  g1[3] = (int)(((rows >> 16) & 0xFFFFu)             // tensor_dim1 hi16
              | (64u << 16));                        // tile_dim0 = 64
  g1[4] = (int)(rows & 0xFFFFu);                     // tile_dim1; tile_dim2=0
  g1[5] = (int)KD;                                   // tensor_dim0_stride lo32
  g1[6] = 0;                                         // stride hi / dim1_stride
  g1[7] = 0;
  i32x4 z4 = {0, 0, 0, 0};
#if __clang_major__ >= 23
  i32x8 z8 = {0, 0, 0, 0, 0, 0, 0, 0};
  __builtin_amdgcn_tensor_load_to_lds(g0, g1, z4, z4, z8, 0);
#else
  __builtin_amdgcn_tensor_load_to_lds(g0, g1, z4, z4, 0);
#endif
}
#endif

// -------------------------- WMMA GEMM --------------------------------------

__global__ __launch_bounds__(256)
void bitnet_wmma_gemm(const __bf16* __restrict__ Aq,   // [MD][KD] bf16
                      const __bf16* __restrict__ Bq,   // [ND][KD] bf16 ternary
                      const float* __restrict__ alphap,
                      const float* __restrict__ bias,
                      float* __restrict__ out) {       // [MD][ND] f32
#ifdef USE_TDM
  __shared__ __bf16 sA[2][BM * LDT];   // 2 x 18,432 B
  __shared__ __bf16 sB[2][BN * LDT];   // 2 x 36,864 B   (total 110.6 KB)
#else
  __shared__ __bf16 sA[1][BM * LDT];
  __shared__ __bf16 sB[1][BN * LDT];
#endif

  const int tid   = threadIdx.x;
  const int lane  = tid & 31;
  const int wave  = tid >> 5;       // 8 waves: 2 in M x 4 in N
  const int waveM = wave & 1;       // wave tile = 64(M) x 64(N)
  const int waveN = wave >> 1;
  const int lrow  = lane & 15;
  const int lhalf = lane >> 4;

  const size_t bm = (size_t)blockIdx.y * BM;
  const size_t bn = (size_t)blockIdx.x * BN;

  const __bf16* aG = Aq + bm * (size_t)KD;
  const __bf16* bG = Bq + bn * (size_t)KD;

  v8f acc[4][4];
#pragma unroll
  for (int mi = 0; mi < 4; ++mi)
#pragma unroll
    for (int ni = 0; ni < 4; ++ni)
      acc[mi][ni] = (v8f)(0.0f);

#ifdef USE_TDM
  // --- TDM double-buffered pipeline: each wave loads its slice -------------
  // A slice: 16 rows x 64 ; B slice: 32 rows x 64 (2 descriptors per wave).
  const int wuni = __builtin_amdgcn_readfirstlane(wave);
  const unsigned long long aSliceG =
      (unsigned long long)(uintptr_t)(aG + (size_t)(wuni * 16) * KD);
  const unsigned long long bSliceG =
      (unsigned long long)(uintptr_t)(bG + (size_t)(wuni * 32) * KD);
  const unsigned ldsA0 = (unsigned)(uintptr_t)&sA[0][0] + (unsigned)(wuni * 16 * LDT * 2);
  const unsigned ldsA1 = (unsigned)(uintptr_t)&sA[1][0] + (unsigned)(wuni * 16 * LDT * 2);
  const unsigned ldsB0 = (unsigned)(uintptr_t)&sB[0][0] + (unsigned)(wuni * 32 * LDT * 2);
  const unsigned ldsB1 = (unsigned)(uintptr_t)&sB[1][0] + (unsigned)(wuni * 32 * LDT * 2);

  tdm_load_2d(aSliceG, ldsA0, 16u);        // tile 0 -> buf 0
  tdm_load_2d(bSliceG, ldsB0, 32u);

  int buf = 0;
  for (int kt = 0; kt < NT; ++kt) {
    __syncthreads();                       // readers of buf^1 (iter kt-1) done
    if (kt + 1 < NT) {
      const unsigned long long ko = (unsigned long long)(kt + 1) * (BKT * 2);
      tdm_load_2d(aSliceG + ko, buf ? ldsA0 : ldsA1, 16u);
      tdm_load_2d(bSliceG + ko, buf ? ldsB0 : ldsB1, 32u);
      __builtin_amdgcn_s_wait_tensorcnt(2);  // my tile-kt slices complete
    } else {
      __builtin_amdgcn_s_wait_tensorcnt(0);
    }
    __syncthreads();                       // all waves' tile-kt slices visible

    const __bf16* As = &sA[buf][0];
    const __bf16* Bs = &sB[buf][0];
#pragma unroll
    for (int kk = 0; kk < BKT; kk += 32) {
      v16bf af[4], bfrag[4];
#pragma unroll
      for (int mi = 0; mi < 4; ++mi)
        af[mi] = frag16(&As[(waveM * 64 + mi * 16 + lrow) * LDT + kk + lhalf * 8], 16);
#pragma unroll
      for (int ni = 0; ni < 4; ++ni)
        bfrag[ni] = frag16(&Bs[(waveN * 64 + ni * 16 + lrow) * LDT + kk + lhalf * 8], 16);
#pragma unroll
      for (int mi = 0; mi < 4; ++mi)
#pragma unroll
        for (int ni = 0; ni < 4; ++ni)
          acc[mi][ni] = __builtin_amdgcn_wmma_f32_16x16x32_bf16(
              false, af[mi], false, bfrag[ni], (short)0, acc[mi][ni], false, false);
    }
    buf ^= 1;
  }
#else
  // --- fallback: cooperative register-staged copy, single buffer -----------
  int grow[12], gcol[12];
#pragma unroll
  for (int i = 0; i < 12; ++i) {
    int chunk = tid + i * 256;             // i<4: A (1024 chunks); i>=4: B (2048)
    grow[i] = (i < 4) ? (chunk >> 3) : ((chunk - 1024) >> 3);
    gcol[i] = (chunk & 7) * 8;
  }
  for (int kt = 0; kt < NT; ++kt) {
    const size_t ko = (size_t)kt * BKT;
    uint4 st[12];
#pragma unroll
    for (int i = 0; i < 4; ++i)
      st[i] = *(const uint4*)(aG + (size_t)grow[i] * KD + ko + gcol[i]);
#pragma unroll
    for (int i = 4; i < 12; ++i)
      st[i] = *(const uint4*)(bG + (size_t)grow[i] * KD + ko + gcol[i]);
    __syncthreads();
#pragma unroll
    for (int i = 0; i < 4; ++i)
      *(uint4*)&sA[0][grow[i] * LDT + gcol[i]] = st[i];
#pragma unroll
    for (int i = 4; i < 12; ++i)
      *(uint4*)&sB[0][grow[i] * LDT + gcol[i]] = st[i];
    __syncthreads();
#pragma unroll
    for (int kk = 0; kk < BKT; kk += 32) {
      v16bf af[4], bfrag[4];
#pragma unroll
      for (int mi = 0; mi < 4; ++mi)
        af[mi] = frag16(&sA[0][(waveM * 64 + mi * 16 + lrow) * LDT + kk + lhalf * 8], 16);
#pragma unroll
      for (int ni = 0; ni < 4; ++ni)
        bfrag[ni] = frag16(&sB[0][(waveN * 64 + ni * 16 + lrow) * LDT + kk + lhalf * 8], 16);
#pragma unroll
      for (int mi = 0; mi < 4; ++mi)
#pragma unroll
        for (int ni = 0; ni < 4; ++ni)
          acc[mi][ni] = __builtin_amdgcn_wmma_f32_16x16x32_bf16(
              false, af[mi], false, bfrag[ni], (short)0, acc[mi][ni], false, false);
    }
  }
#endif

  // Epilogue: C/D layout — VGPR r: M = base + lhalf*8 + r, N = lane%16
  const float alpha = alphap[0];
#pragma unroll
  for (int ni = 0; ni < 4; ++ni) {
    const size_t n = bn + (size_t)(waveN * 64 + ni * 16 + lrow);
    const float bb = bias[n];
#pragma unroll
    for (int mi = 0; mi < 4; ++mi) {
      const size_t m0 = bm + (size_t)(waveM * 64 + mi * 16 + lhalf * 8);
#pragma unroll
      for (int r = 0; r < 8; ++r)
        out[(m0 + r) * (size_t)ND + n] = alpha * acc[mi][ni][r] + bb;
    }
  }
}

// -------------------------- launch -----------------------------------------

extern "C" void kernel_launch(void* const* d_in, const int* in_sizes, int n_in,
                              void* d_out, int out_size, void* d_ws, size_t ws_size,
                              hipStream_t stream) {
  (void)in_sizes; (void)n_in; (void)out_size; (void)ws_size;

  const float* x     = (const float*)d_in[0];  // [2,2048,4096]
  const float* w     = (const float*)d_in[1];  // [11008,4096]
  const float* alpha = (const float*)d_in[2];  // [1]
  const float* bias  = (const float*)d_in[3];  // [11008]
  float* out = (float*)d_out;                  // [2,2048,11008]

  __bf16* xq = (__bf16*)d_ws;                  // MD*KD bf16 (33.5 MB)
  __bf16* wq = xq + (size_t)MD * KD;           // ND*KD bf16 (90.2 MB)

  const int nx8 = (MD * KD) / 8;   // 2,097,152
  const int nw8 = (ND * KD) / 8;   // 5,636,096
  cvt_x_bf16<<<(nx8 + 255) / 256, 256, 0, stream>>>(x, xq, nx8);
  quant_w_ternary<<<(nw8 + 255) / 256, 256, 0, stream>>>(w, alpha, wq, nw8);

  dim3 grid(ND / BN, MD / BM);     // (43, 32)
  bitnet_wmma_gemm<<<grid, 256, 0, stream>>>(xq, wq, alpha, bias, out);
}
